// MultiHeadAttention_20194936226332
// MI455X (gfx1250) — matrix operand order repaired and checked by reading, not compiled, not run on hardware
//
#include <hip/hip_runtime.h>

// ---------------------------------------------------------------------------
// MHA forward for MI455X (gfx1250, wave32, WMMA).
// fp32 inputs are converted to bf16; all three matmul stages (QKV projection,
// attention, output projection) run on v_wmma_f32_16x16x32_bf16 with fp32
// accumulation. bf16 tile staging uses the CDNA5 async Global->LDS engine
// (global_load_async_to_lds_b128 + s_wait_asynccnt).
// ---------------------------------------------------------------------------

typedef __attribute__((ext_vector_type(16))) __bf16 v16bf;
typedef __attribute__((ext_vector_type(8)))  float  v8f;

union Frag {
    v16bf bf;
    uint4 u4[2];
};

__device__ __forceinline__ v8f wmma_bf16(const Frag& a, const Frag& b, v8f c) {
    return __builtin_amdgcn_wmma_f32_16x16x32_bf16(
        /*neg_a=*/false, a.bf, /*neg_b=*/false, b.bf,
        /*c_mod=*/(short)0, c, /*reuse_a=*/false, /*reuse_b=*/false);
}

// Async copy of 16 B from global memory directly into LDS (ASYNCcnt-tracked).
// Low 32 bits of a generic pointer to a __shared__ object are the LDS byte
// address (flat LDS aperture maps addr[31:0] -> LDS).
__device__ __forceinline__ void async_b128(void* lds_dst, const void* gsrc) {
    const unsigned lds = (unsigned)(uintptr_t)lds_dst;
    asm volatile("global_load_async_to_lds_b128 %0, %1, off"
                 :: "v"(lds), "v"(gsrc)
                 : "memory");
}
__device__ __forceinline__ void wait_async0() {
    asm volatile("s_wait_asynccnt 0x0" ::: "memory");
}

// ===========================================================================
// Kernel 1: QKV projection. X[4096,1024] fp32  x  W[1024,3072] fp32 + bias.
// Output bf16: Q,K in [B=2,H=16,T=2048,hd=64]; V pre-transposed in
// [B=2,H=16,hd=64,T=2048] so the attention kernel can async-copy V tiles
// straight into its [d][key] LDS layout.
// Tile: 128(M) x 128(N) x 32(K); 8 waves, each a 2x4 grid of 16x16 WMMAs.
// ===========================================================================
__global__ __launch_bounds__(256) void qkv_gemm(
    const float* __restrict__ X, const float* __restrict__ W,
    const float* __restrict__ bias,
    __bf16* __restrict__ q_ws, __bf16* __restrict__ k_ws,
    __bf16* __restrict__ vt_ws)
{
    // row stride 40 bf16 = 80 B: multiple of 16 B (aligned b128) and 20 banks
    // (16 consecutive rows hit distinct banks).
    __shared__ __align__(16) __bf16 As[128][40];  // [m][k]
    __shared__ __align__(16) __bf16 Bs[128][40];  // [n][k] (transposed)

    const int tid  = threadIdx.x;
    const int lane = tid & 31;
    const int wave = tid >> 5;
    const int m0 = blockIdx.y * 128;           // M = 4096 -> gridDim.y = 32
    const int n0 = blockIdx.x * 128;           // N = 3072 -> gridDim.x = 24
    const int wm = (wave & 3) * 32;
    const int wn = (wave >> 2) * 64;
    const int mrow = lane & 15;
    const int hi   = lane >> 4;

    v8f acc[2][4] = {};

    for (int kk = 0; kk < 1024; kk += 32) {
        __syncthreads();
        // A tile: 128 rows x 32 k, fp32 -> bf16 (coalesced float4 reads)
        #pragma unroll
        for (int i = 0; i < 4; ++i) {
            const int idx = tid + i * 256;               // 0..1023
            const int row = idx >> 3, c4 = (idx & 7) * 4;
            const float4 f = *(const float4*)&X[(size_t)(m0 + row) * 1024 + kk + c4];
            As[row][c4 + 0] = (__bf16)f.x; As[row][c4 + 1] = (__bf16)f.y;
            As[row][c4 + 2] = (__bf16)f.z; As[row][c4 + 3] = (__bf16)f.w;
        }
        // B tile: W[k][n] -> Bs[n][k] transposed, fp32 -> bf16
        #pragma unroll
        for (int i = 0; i < 4; ++i) {
            const int idx = tid + i * 256;               // 32 k-rows x 32 n-float4
            const int kr = idx >> 5, n4 = (idx & 31) * 4;
            const float4 f = *(const float4*)&W[(size_t)(kk + kr) * 3072 + n0 + n4];
            Bs[n4 + 0][kr] = (__bf16)f.x; Bs[n4 + 1][kr] = (__bf16)f.y;
            Bs[n4 + 2][kr] = (__bf16)f.z; Bs[n4 + 3][kr] = (__bf16)f.w;
        }
        __syncthreads();

        Frag a[2], b[4];
        #pragma unroll
        for (int mt = 0; mt < 2; ++mt) {
            // A-frag: lane half selects K 0-15 vs 16-31
            const __bf16* p = &As[wm + mt * 16 + mrow][hi * 16];
            a[mt].u4[0] = *(const uint4*)p;
            a[mt].u4[1] = *(const uint4*)(p + 8);
        }
        #pragma unroll
        for (int nt = 0; nt < 4; ++nt) {
            // B-frag: lane half selects K 0-15 vs 16-31 (contiguous in Bs row)
            const __bf16* p = &Bs[wn + nt * 16 + mrow][hi * 16];
            b[nt].u4[0] = *(const uint4*)p;
            b[nt].u4[1] = *(const uint4*)(p + 8);
        }
        #pragma unroll
        for (int mt = 0; mt < 2; ++mt)
            #pragma unroll
            for (int nt = 0; nt < 4; ++nt)
                acc[mt][nt] = wmma_bf16(a[mt], b[nt], acc[mt][nt]);
    }

    // Epilogue: add bias, scatter to Q/K [B,H,T,64] and V^T [B,H,64,T].
    // The 16-column range of one n-tile is 16-aligned, so sel/h/dbase are
    // wave-uniform: force them scalar so routing is s_cbranch, not exec-mask.
    #pragma unroll
    for (int nt = 0; nt < 4; ++nt) {
        const int scol  = __builtin_amdgcn_readfirstlane(n0 + wn + nt * 16);
        const int sel   = scol >> 10;              // 0=Q 1=K 2=V (scalar)
        const int h     = (scol & 1023) >> 6;      // scalar
        const int dbase = scol & 63;               // scalar
        const int d     = dbase + mrow;
        const float bv  = bias[scol + mrow];
        #pragma unroll
        for (int mt = 0; mt < 2; ++mt) {
            const int row0 = m0 + wm + mt * 16 + hi * 8;   // 8-aligned
            const int bidx = row0 >> 11;                   // uniform over r
            const int t0   = row0 & 2047;
            const size_t bhh = (size_t)bidx * 16 + h;
            if (sel == 0) {
                __bf16* dst = q_ws + (bhh * 2048 + t0) * 64 + d;
                #pragma unroll
                for (int r = 0; r < 8; ++r)
                    dst[r * 64] = (__bf16)(acc[mt][nt][r] + bv);
            } else if (sel == 1) {
                __bf16* dst = k_ws + (bhh * 2048 + t0) * 64 + d;
                #pragma unroll
                for (int r = 0; r < 8; ++r)
                    dst[r * 64] = (__bf16)(acc[mt][nt][r] + bv);
            } else {
                __bf16* dst = vt_ws + (bhh * 64 + d) * 2048 + t0;
                #pragma unroll
                for (int r = 0; r < 8; ++r)
                    dst[r] = (__bf16)(acc[mt][nt][r] + bv);
            }
        }
    }
}

// ===========================================================================
// Kernel 2: causal flash attention.
// Grid: x = query block (128 rows, 16 blocks), y = b*H (32).
// 8 waves/WG; wave w owns query rows q0 + 16w .. +15. Loop over 32-key
// blocks; K ([key][d]) and V^T ([d][key]) tiles staged with async
// Global->LDS b128 copies.
// scale = 1/sqrt(3*1024/16) = 1/sqrt(192)   (nonstandard, per reference)
// ===========================================================================
__global__ __launch_bounds__(256) void flash_attn(
    const __bf16* __restrict__ q_ws, const __bf16* __restrict__ k_ws,
    const __bf16* __restrict__ vt_ws, __bf16* __restrict__ y_ws)
{
    __shared__ __align__(16) __bf16 Ksm[32][72];     // [key][d], 144 B rows
    __shared__ __align__(16) __bf16 Vsm[64][40];     // [d][key], 80 B rows
    __shared__ __align__(16) __bf16 Psm[8][16][40];  // per-wave P relayout

    const int tid  = threadIdx.x;
    const int lane = tid & 31;
    const int wave = tid >> 5;
    const int bh = blockIdx.y;                 // 0..31
    const int qb = blockIdx.x;                 // 0..15
    const int q0 = qb * 128;
    const int mrow = lane & 15;
    const int hi   = lane >> 4;
    const int qrow_base = q0 + wave * 16;

    const size_t head_off = (size_t)bh * 2048 * 64;
    const __bf16* Qp  = q_ws  + head_off;
    const __bf16* Kp  = k_ws  + head_off;
    const __bf16* Vtp = vt_ws + head_off;      // [64][2048]

    // Load the wave's Q fragments once (A layout, head_dim split into 2 WMMA-K).
    Frag qf[2];
    {
        const __bf16* qrow = Qp + (size_t)(qrow_base + mrow) * 64;
        #pragma unroll
        for (int dh = 0; dh < 2; ++dh) {
            qf[dh].u4[0] = *(const uint4*)(qrow + dh * 32 + hi * 16);
            qf[dh].u4[1] = *(const uint4*)(qrow + dh * 32 + hi * 16 + 8);
        }
    }

    v8f O[4] = {};
    float m_i[8], l_i[8];
    #pragma unroll
    for (int r = 0; r < 8; ++r) { m_i[r] = -1e30f; l_i[r] = 0.0f; }

    const float scale = 0.07216878364870323f;  // 1/sqrt(192)
    const int nkb = qb * 4 + 4;                // causal: keys up to q0+127

    // Per-thread staging pointers, advanced incrementally per key block.
    const int krow = tid >> 3, kc = (tid & 7) * 8;   // K: 32 rows x 8 chunks
    const int vrow = tid >> 2, vc = (tid & 3) * 8;   // V: 64 rows x 4 chunks
    const __bf16* gK = Kp  + (size_t)krow * 64 + kc;
    const __bf16* gV = Vtp + (size_t)vrow * 2048 + vc;
    void* ldsK = &Ksm[krow][kc];
    void* ldsV = &Vsm[vrow][vc];

    for (int kb = 0; kb < nkb; ++kb) {
        const int j0 = kb * 32;
        __syncthreads();
        // Async-stage K block [32][64] and V^T block [64][32]: 16 B per thread
        // each, VGPR-free, tracked on ASYNCcnt.
        async_b128(ldsK, gK);
        async_b128(ldsV, gV);
        gK += 32 * 64;                  // next 32 keys (row-major [key][d])
        gV += 32;                       // next 32 keys (row-major [d][key])
        wait_async0();
        __syncthreads();

        // S = Q * K^T  (two 16x16 key sub-tiles, contraction over hd=64)
        v8f S[2];
        #pragma unroll
        for (int s = 0; s < 2; ++s) {
            v8f sa = {};
            #pragma unroll
            for (int dh = 0; dh < 2; ++dh) {
                Frag kf;
                const __bf16* kp = &Ksm[s * 16 + mrow][dh * 32 + hi * 16];
                kf.u4[0] = *(const uint4*)kp;
                kf.u4[1] = *(const uint4*)(kp + 8);
                sa = wmma_bf16(qf[dh], kf, sa);
            }
            S[s] = sa;
        }

        // Scale + causal mask + online softmax (row groups of 16 lanes).
        float p[2][8];
        #pragma unroll
        for (int r = 0; r < 8; ++r) {
            const int qrow = qrow_base + r + hi * 8;
            float v0 = S[0][r] * scale; if (j0 + mrow      > qrow) v0 = -1e30f;
            float v1 = S[1][r] * scale; if (j0 + 16 + mrow > qrow) v1 = -1e30f;
            p[0][r] = v0; p[1][r] = v1;
            float mx = fmaxf(v0, v1);
            mx = fmaxf(mx, __shfl_xor(mx, 1, 16));
            mx = fmaxf(mx, __shfl_xor(mx, 2, 16));
            mx = fmaxf(mx, __shfl_xor(mx, 4, 16));
            mx = fmaxf(mx, __shfl_xor(mx, 8, 16));
            const float mnew  = fmaxf(m_i[r], mx);
            const float alpha = __expf(m_i[r] - mnew);
            const float e0 = __expf(p[0][r] - mnew);
            const float e1 = __expf(p[1][r] - mnew);
            p[0][r] = e0; p[1][r] = e1;
            float rs = e0 + e1;
            rs += __shfl_xor(rs, 1, 16);
            rs += __shfl_xor(rs, 2, 16);
            rs += __shfl_xor(rs, 4, 16);
            rs += __shfl_xor(rs, 8, 16);
            l_i[r] = l_i[r] * alpha + rs;
            m_i[r] = mnew;
            #pragma unroll
            for (int nt = 0; nt < 4; ++nt) O[nt][r] *= alpha;
        }

        // Relayout P (C-layout) -> A-fragment layout through per-wave LDS.
        #pragma unroll
        for (int s = 0; s < 2; ++s)
            #pragma unroll
            for (int r = 0; r < 8; ++r)
                Psm[wave][r + hi * 8][s * 16 + mrow] = (__bf16)p[s][r];
        // Same-wave LDS ops are kept in order; compiler inserts s_wait_dscnt.
        Frag pf;
        {
            const __bf16* pp = &Psm[wave][mrow][hi * 16];
            pf.u4[0] = *(const uint4*)pp;
            pf.u4[1] = *(const uint4*)(pp + 8);
        }
        // O += P * V  (K=32 keys, four 16-wide d tiles)
        #pragma unroll
        for (int nt = 0; nt < 4; ++nt) {
            Frag vf;
            const __bf16* vp = &Vsm[nt * 16 + mrow][hi * 16];
            vf.u4[0] = *(const uint4*)vp;
            vf.u4[1] = *(const uint4*)(vp + 8);
            O[nt] = wmma_bf16(pf, vf, O[nt]);
        }
    }

    // Normalize and write y as bf16 [B, T, C] (head-interleaved columns).
    const int b = bh >> 4, h = bh & 15;
    #pragma unroll
    for (int r = 0; r < 8; ++r) {
        const float inv = 1.0f / l_i[r];
        const int qrow = qrow_base + r + hi * 8;
        __bf16* dst = y_ws + ((size_t)b * 2048 + qrow) * 1024 + h * 64;
        #pragma unroll
        for (int nt = 0; nt < 4; ++nt)
            dst[nt * 16 + mrow] = (__bf16)(O[nt][r] * inv);
    }
}

// ===========================================================================
// Kernel 3: output projection. y[4096,1024] bf16 x W_proj[1024,1024] fp32.
// A tile staged with async Global->LDS copies (already bf16); B converted.
// ===========================================================================
__global__ __launch_bounds__(256) void proj_gemm(
    const __bf16* __restrict__ Y, const float* __restrict__ W,
    const float* __restrict__ bias, float* __restrict__ out)
{
    __shared__ __align__(16) __bf16 As[128][40];
    __shared__ __align__(16) __bf16 Bs[128][40];

    const int tid  = threadIdx.x;
    const int lane = tid & 31;
    const int wave = tid >> 5;
    const int m0 = blockIdx.y * 128;           // M = 4096 -> gridDim.y = 32
    const int n0 = blockIdx.x * 128;           // N = 1024 -> gridDim.x = 8
    const int wm = (wave & 3) * 32;
    const int wn = (wave >> 2) * 64;
    const int mrow = lane & 15;
    const int hi   = lane >> 4;

    v8f acc[2][4] = {};

    // Per-thread async A-staging coordinates.
    const int arow0 = tid >> 2,         ac0 = (tid & 3) * 8;   // i = 0
    const int arow1 = (tid + 256) >> 2, ac1 = (tid & 3) * 8;   // i = 1

    for (int kk = 0; kk < 1024; kk += 32) {
        __syncthreads();
        // A tile already bf16: async Global->LDS, 2 x 16 B per thread.
        async_b128(&As[arow0][ac0], Y + (size_t)(m0 + arow0) * 1024 + kk + ac0);
        async_b128(&As[arow1][ac1], Y + (size_t)(m0 + arow1) * 1024 + kk + ac1);
        // B tile: W[k][n] -> Bs[n][k], fp32 -> bf16.
        #pragma unroll
        for (int i = 0; i < 4; ++i) {
            const int idx = tid + i * 256;
            const int kr = idx >> 5, n4 = (idx & 31) * 4;
            const float4 f = *(const float4*)&W[(size_t)(kk + kr) * 1024 + n0 + n4];
            Bs[n4 + 0][kr] = (__bf16)f.x; Bs[n4 + 1][kr] = (__bf16)f.y;
            Bs[n4 + 2][kr] = (__bf16)f.z; Bs[n4 + 3][kr] = (__bf16)f.w;
        }
        wait_async0();
        __syncthreads();

        Frag a[2], b[4];
        #pragma unroll
        for (int mt = 0; mt < 2; ++mt) {
            const __bf16* p = &As[wm + mt * 16 + mrow][hi * 16];
            a[mt].u4[0] = *(const uint4*)p;
            a[mt].u4[1] = *(const uint4*)(p + 8);
        }
        #pragma unroll
        for (int nt = 0; nt < 4; ++nt) {
            const __bf16* p = &Bs[wn + nt * 16 + mrow][hi * 16];
            b[nt].u4[0] = *(const uint4*)p;
            b[nt].u4[1] = *(const uint4*)(p + 8);
        }
        #pragma unroll
        for (int mt = 0; mt < 2; ++mt)
            #pragma unroll
            for (int nt = 0; nt < 4; ++nt)
                acc[mt][nt] = wmma_bf16(a[mt], b[nt], acc[mt][nt]);
    }

    #pragma unroll
    for (int mt = 0; mt < 2; ++mt) {
        #pragma unroll
        for (int nt = 0; nt < 4; ++nt) {
            const int col = n0 + wn + nt * 16 + mrow;
            const float bv = bias[col];
            #pragma unroll
            for (int r = 0; r < 8; ++r) {
                const int row = m0 + wm + mt * 16 + r + hi * 8;
                out[(size_t)row * 1024 + col] = acc[mt][nt][r] + bv;
            }
        }
    }
}

// ===========================================================================
// Launcher
// ===========================================================================
extern "C" void kernel_launch(void* const* d_in, const int* in_sizes, int n_in,
                              void* d_out, int out_size, void* d_ws, size_t ws_size,
                              hipStream_t stream) {
    const float* x      = (const float*)d_in[0];   // [2,2048,1024]
    const float* W_attn = (const float*)d_in[1];   // [1024,3072]
    const float* b_attn = (const float*)d_in[2];   // [3072]
    const float* W_proj = (const float*)d_in[3];   // [1024,1024]
    const float* b_proj = (const float*)d_in[4];   // [1024]
    float* out = (float*)d_out;                    // [2,2048,1024] fp32

    // Workspace (bf16): Q [B,H,T,64] | K [B,H,T,64] | V^T [B,H,64,T] | Y [B,T,C]
    const size_t HT = (size_t)2 * 16 * 2048 * 64;  // 4,194,304 elements
    __bf16* ws    = (__bf16*)d_ws;
    __bf16* q_ws  = ws;
    __bf16* k_ws  = ws + HT;
    __bf16* vt_ws = ws + 2 * HT;
    __bf16* y_ws  = ws + 3 * HT;

    qkv_gemm <<<dim3(24, 32), 256, 0, stream>>>(x, W_attn, b_attn, q_ws, k_ws, vt_ws);
    flash_attn<<<dim3(16, 32), 256, 0, stream>>>(q_ws, k_ws, vt_ws, y_ws);
    proj_gemm <<<dim3(8, 32),  256, 0, stream>>>(y_ws, W_proj, b_proj, out);
}
